// SimAttention_88630945120841
// MI455X (gfx1250) — matrix operand order repaired and checked by reading, not compiled, not run on hardware
//
#include <hip/hip_runtime.h>
#include <hip/hip_bf16.h>
#include <cstdint>

// ---------------------------------------------------------------------------
// Types for CDNA5 WMMA (wave32): A/B = 16 bf16 per lane, C/D = 8 f32 per lane
// ---------------------------------------------------------------------------
typedef __attribute__((ext_vector_type(16))) __bf16 v16bf;
typedef __attribute__((ext_vector_type(8)))  float  v8f;
// Tensor-DMA descriptor vectors
typedef __attribute__((ext_vector_type(4))) unsigned int u32x4;
typedef __attribute__((ext_vector_type(8))) int          i32x8;
typedef __attribute__((ext_vector_type(4))) int          i32x4;

union FragBF {
  v16bf    v;
  uint32_t u32[8];
  uint64_t u64[4];
};

__device__ __forceinline__ float bf2f(__bf16 b) {
  uint16_t h; __builtin_memcpy(&h, &b, 2);
  uint32_t u = ((uint32_t)h) << 16;
  float f; __builtin_memcpy(&f, &u, 4);
  return f;
}

__device__ __forceinline__ __bf16 f2bf(float f) {
  uint32_t u; __builtin_memcpy(&u, &f, 4);
  uint32_t r = u + 0x7FFFu + ((u >> 16) & 1u);   // round-to-nearest-even
  uint16_t h = (uint16_t)(r >> 16);
  __bf16 b; __builtin_memcpy(&b, &h, 2);
  return b;
}

// TDM issue helper — builtin arity differs across toolchains (bridge doc):
// ROCm 7.2 / clang-22: 5 args; amdgpu-toolchain / clang-23: 6 args.
__device__ __forceinline__ void tdm_load(u32x4 g0, i32x8 g1, i32x4 g2, i32x4 g3) {
#if defined(__clang_major__) && (__clang_major__ >= 23)
  i32x8 z = {0, 0, 0, 0, 0, 0, 0, 0};
  __builtin_amdgcn_tensor_load_to_lds(g0, g1, g2, g3, z, 0);
#else
  __builtin_amdgcn_tensor_load_to_lds(g0, g1, g2, g3, 0);
#endif
}

// ---------------------------------------------------------------------------
// f32 -> bf16 conversion
// ---------------------------------------------------------------------------
__global__ void cvt_f32_bf16_kernel(const float* __restrict__ in,
                                    __bf16* __restrict__ out, int n) {
  int i = blockIdx.x * 256 + threadIdx.x;
  if (i < n) out[i] = f2bf(in[i]);
}

// ---------------------------------------------------------------------------
// LDS fragment loaders per CDNA5 ISA 7.12.2 layouts
// A (16x32 bf16, MxK): lane l -> row M = l&15; element pair j (VGPR j):
//   K = koff + (j>=4 ? 16 : 0) + ((l>>4)*8) + 2*(j&3) + {0,1}
// B (32x16 bf16, KxN) staged TRANSPOSED in LDS as [n][k]:
//   lane l -> col N = l&15, K same as A: koff + (j>=4 ? 16 : 0) + ((l>>4)*8) + 2*(j&3) + {0,1}
// ---------------------------------------------------------------------------
#define ASTRIDE 80   // 64 + 16 bf16 pad (row stride 160 B; TDM pad reproduces)
#define BSTRIDE 80

__device__ __forceinline__ void loadAFrag(FragBF& f, const __bf16* ldsA,
                                          int mBase, int lane, int koff) {
  const __bf16* rp = ldsA + (mBase + (lane & 15)) * ASTRIDE + koff;
  int hi = (lane >> 4) * 8;
#pragma unroll
  for (int j = 0; j < 8; ++j) {
    int k = ((j >> 2) * 16) + hi + ((j & 3) * 2);
    f.u32[j] = *(const uint32_t*)(rp + k);
  }
}

__device__ __forceinline__ void loadBFrag(FragBF& f, const __bf16* ldsB,
                                          int nBase, int lane, int koff) {
  const __bf16* rp = ldsB + (nBase + (lane & 15)) * BSTRIDE + koff +
                     ((lane >> 4) * 8);
#pragma unroll
  for (int j = 0; j < 8; ++j) f.u32[j] = *(const uint32_t*)(rp + ((j >> 2) * 16) + ((j & 3) * 2));
}

// ---------------------------------------------------------------------------
// Tiled bf16 WMMA GEMM: C = relu(A @ W + bias)
//   A: M x K bf16, W: K x N bf16.  Block: 256 thr = 8 waves,
//   tile 128(M) x 64(N), BK = 64 -> 8 WMMAs per barrier round.
//   GATHER=false: A tile staged by the Tensor Data Mover (tensor_load_to_lds
//     with LDS padding 8 DW per 32 DW => ASTRIDE=80), TENSORcnt-tracked.
//   GATHER=true : A rows permuted via rowIdx, staged manually (b128 -> b128).
// ---------------------------------------------------------------------------
template <bool GATHER, bool WB, bool WF>
__global__ __launch_bounds__(256)
void gemm_bf16_relu_kernel(const __bf16* __restrict__ A,
                           const __bf16* __restrict__ W,
                           const float* __restrict__ bias,
                           const int* __restrict__ rowIdx,
                           __bf16* __restrict__ outB,
                           float* __restrict__ outF,
                           int M, int N, int K) {
  // Single carved LDS block so ldsA sits at LDS offset 0 (TDM D#.lds_addr = 0)
  __shared__ __align__(16) char smem[(128 * ASTRIDE + 64 * BSTRIDE) * 2];
  __bf16* ldsA = (__bf16*)smem;
  __bf16* ldsB = (__bf16*)(smem + 128 * ASTRIDE * 2);

  const int tid  = threadIdx.x;
  const int lane = tid & 31;
  const int wave = tid >> 5;
  const int wm   = wave & 3;        // 4 waves along M (4*32 = 128)
  const int wn   = wave >> 2;       // 2 waves along N (2*32 = 64)
  const int mBlock = blockIdx.y * 128;
  const int nBlock = blockIdx.x * 64;

  // Manual A staging geometry (gather path): 4 passes, 32 rows, b128 loads
  const int aRow = tid >> 3;            // 0..31
  const int aCol = (tid & 7) * 8;       // 0..56
  int arows[4];
  if constexpr (GATHER) {
#pragma unroll
    for (int p = 0; p < 4; ++p) arows[p] = rowIdx[mBlock + p * 32 + aRow];
  } else {
    (void)arows;
  }

  v8f acc[4];
#pragma unroll
  for (int t = 0; t < 4; ++t)
#pragma unroll
    for (int e = 0; e < 8; ++e) acc[t][e] = 0.0f;

  for (int kb = 0; kb < K; kb += 64) {
    if constexpr (!GATHER) {
      // --- TDM: DMA the 128x64 bf16 A tile into padded LDS rows ------------
      if (wave == 0) {
        uint64_t ga = (uint64_t)(uintptr_t)(A + (size_t)mBlock * K + kb);
        u32x4 g0 = { 1u,                                   // count=1 (valid)
                     0u,                                   // lds_addr = 0
                     (uint32_t)ga,                         // global_addr[31:0]
                     (uint32_t)((ga >> 32) & 0x1FFFFFFu) | (2u << 30) }; // +type=2
        // d0: data_size=1 (2B) | pad_enable | pad_interval=4 (32 DW) |
        //     pad_amount=7 (8 DW)  -> LDS row stride = 160 B = ASTRIDE
        uint32_t d0 = (1u << 16) | (1u << 20) | (4u << 22) | (7u << 25);
        uint32_t td0 = (uint32_t)K;      // tensor_dim0 (elements)
        uint32_t td1 = (uint32_t)M;      // tensor_dim1 (rows)
        i32x8 g1 = { (int)d0,
                     (int)((td0 & 0xFFFFu) << 16),                       // dim0[15:0]
                     (int)(((td0 >> 16) & 0xFFFFu) | ((td1 & 0xFFFFu) << 16)),
                     (int)(((td1 >> 16) & 0xFFFFu) | (64u << 16)),       // tile_dim0=64
                     (int)128u,                                          // tile_dim1=128
                     (int)(uint32_t)K,                                   // dim0_stride lo
                     0, 0 };
        i32x4 g2 = { 1, 0, 0, 0 };       // tensor_dim2=1, rest unused (2D)
        i32x4 g3 = { 0, 0, 0, 0 };
        tdm_load(g0, g1, g2, g3);
      }
    } else {
      // --- Manual gathered A staging: global b128 -> ds b128 ---------------
#pragma unroll
      for (int p = 0; p < 4; ++p) {
        int r = p * 32 + aRow;
        uint4 v = *(const uint4*)(A + (size_t)arows[p] * K + kb + aCol);
        *(uint4*)(ldsA + r * ASTRIDE + aCol) = v;
      }
    }

    // Stage B tile transposed: global W[k][n] -> ldsB[n][k]
#pragma unroll
    for (int p = 0; p < 4; ++p) {
      int kk = p * 16 + (tid >> 4);
      int n0 = (tid & 15) * 4;
      union { uint64_t u; __bf16 e[4]; } t;
      t.u = *(const uint64_t*)(W + (size_t)(kb + kk) * N + nBlock + n0);
#pragma unroll
      for (int i = 0; i < 4; ++i) ldsB[(n0 + i) * BSTRIDE + kk] = t.e[i];
    }
    if (kb + 64 < K)   // warm near caches for next W tile
      __builtin_prefetch(W + (size_t)(kb + 64 + (tid >> 4)) * N + nBlock, 0, 3);

    if constexpr (!GATHER) {
      if (wave == 0) __builtin_amdgcn_s_wait_tensorcnt(0);  // TDM tile landed
    }
    __syncthreads();

#pragma unroll
    for (int s = 0; s < 2; ++s) {       // two 32-wide WMMA K steps per tile
      FragBF a0, a1, b0, b1;
      loadAFrag(a0, ldsA, wm * 32,      lane, s * 32);
      loadAFrag(a1, ldsA, wm * 32 + 16, lane, s * 32);
      loadBFrag(b0, ldsB, wn * 32,      lane, s * 32);
      loadBFrag(b1, ldsB, wn * 32 + 16, lane, s * 32);

      acc[0] = __builtin_amdgcn_wmma_f32_16x16x32_bf16(false, a0.v, false, b0.v,
                                                       (short)0, acc[0], false, false);
      acc[1] = __builtin_amdgcn_wmma_f32_16x16x32_bf16(false, a0.v, false, b1.v,
                                                       (short)0, acc[1], false, false);
      acc[2] = __builtin_amdgcn_wmma_f32_16x16x32_bf16(false, a1.v, false, b0.v,
                                                       (short)0, acc[2], false, false);
      acc[3] = __builtin_amdgcn_wmma_f32_16x16x32_bf16(false, a1.v, false, b1.v,
                                                       (short)0, acc[3], false, false);
    }
    __syncthreads();
  }

  // Epilogue: C/D layout -> m = base + r + 8*(lane>=16), n = base + (lane&15)
#pragma unroll
  for (int t = 0; t < 4; ++t) {
    int sm = (t >> 1) * 16;
    int sn = (t & 1) * 16;
#pragma unroll
    for (int r = 0; r < 8; ++r) {
      int m = mBlock + wm * 32 + sm + r + ((lane >> 4) << 3);
      int n = nBlock + wn * 32 + sn + (lane & 15);
      float v = acc[t][r] + bias[n];
      v = v > 0.0f ? v : 0.0f;
      size_t off = (size_t)m * N + n;
      if constexpr (WF) outF[off] = v;
      if constexpr (WB) outB[off] = f2bf(v);
    }
  }
}

// ---------------------------------------------------------------------------
// Per-row (b,n): lsh = sum_h sign(q>0); descending bitonic argsort of D=256;
// positional encoding; EMA-weighted dot -> simv[row]
// ---------------------------------------------------------------------------
__global__ __launch_bounds__(256)
void lsh_simv_kernel(const __bf16* __restrict__ q, float* __restrict__ simv) {
  const int D = 256, H = 8;
  int row = blockIdx.x;
  int d = threadIdx.x;

  const __bf16* qp = q + (size_t)row * (H * D) + d;
  float lsh = 0.0f;
#pragma unroll
  for (int h = 0; h < H; ++h)
    lsh += (bf2f(qp[h * D]) > 0.0f) ? 1.0f : -1.0f;   // 2*min(ceil(relu),1)-1

  __shared__ float sk[256];
  __shared__ int   si[256];
  __shared__ float red[256];
  sk[d] = lsh; si[d] = d;
  __syncthreads();

  // Bitonic sort, descending by key, ties by ascending original index (stable)
  for (int k = 2; k <= 256; k <<= 1) {
    for (int j = k >> 1; j > 0; j >>= 1) {
      int p = d ^ j;
      if (p > d) {
        float ka = sk[d], kb = sk[p];
        int   ia = si[d], ib = si[p];
        bool aFirst = (ka > kb) || (ka == kb && ia < ib);
        bool up = ((d & k) == 0);
        if (up ? !aFirst : aFirst) {
          sk[d] = kb; sk[p] = ka; si[d] = ib; si[p] = ia;
        }
      }
      __syncthreads();
    }
  }

  float ls  = sk[d];
  float pf  = (float)si[d];
  float fr  = expf(-pf * (2.302585093f / 256.0f));     // 10^(-pf/D)
  float par = (si[d] & 1) ? 1.0f : 0.0f;
  float pe  = sinf(fr) * par + cosf(fr) * (1.0f - par);
  float val = ls + pe * 2.5f;                          // pe / EMA_A
  float c   = 0.4f * expf((float)d * -0.5108256238f);  // 0.4 * 0.6^d
  red[d] = val * c;
  __syncthreads();
  for (int s = 128; s > 0; s >>= 1) {
    if (d < s) red[d] += red[d + s];
    __syncthreads();
  }
  if (d == 0) simv[row] = red[0];
}

// ---------------------------------------------------------------------------
// Per-batch ascending bitonic argsort of simv over N=4096 (32 KB LDS).
// Emits global row indices for GEMM2's gathered A-tile loads.
// ---------------------------------------------------------------------------
__global__ __launch_bounds__(1024)
void argsort_n_kernel(const float* __restrict__ simv, int* __restrict__ ordRows) {
  const int N = 4096;
  __shared__ float sk[4096];
  __shared__ int   si[4096];
  int b = blockIdx.x, tid = threadIdx.x;

  for (int i = tid; i < N; i += 1024) { sk[i] = simv[b * N + i]; si[i] = i; }
  __syncthreads();

  for (int k = 2; k <= N; k <<= 1) {
    for (int j = k >> 1; j > 0; j >>= 1) {
      for (int t = tid; t < N / 2; t += 1024) {
        int i = ((t & ~(j - 1)) << 1) | (t & (j - 1));
        int p = i | j;
        bool up = ((i & k) == 0);
        float ka = sk[i], kb = sk[p];
        int   ia = si[i], ib = si[p];
        bool bFirst = (kb < ka) || (kb == ka && ib < ia);  // strict total order
        if (up ? bFirst : !bFirst) {
          sk[i] = kb; sk[p] = ka; si[i] = ib; si[p] = ia;
        }
      }
      __syncthreads();
    }
  }
  for (int i = tid; i < N; i += 1024) ordRows[b * N + i] = b * N + si[i];
}

// ---------------------------------------------------------------------------
// Fused window(9) attention + residual + LayerNorm. One 256-thr block per row.
// ---------------------------------------------------------------------------
__global__ __launch_bounds__(256)
void attn_ln_kernel(const float* __restrict__ q2, const float* __restrict__ vmat,
                    const float* __restrict__ x, const float* __restrict__ gamma,
                    const float* __restrict__ beta, float* __restrict__ out) {
  const int N = 4096, D = 256;
  int row = blockIdx.x;
  int b = row >> 12, n = row & 4095;
  int d = threadIdx.x;

  int start = n - 4;
  if (start < 0) start = 0;
  if (start > N - 9) start = N - 9;

  size_t base = (size_t)row * D;
  float qd = q2[base + d];

  __shared__ float red[256];
  __shared__ float sc[9];
  __shared__ float smu, svar;

  for (int w = 0; w < 9; ++w) {
    size_t kro = (size_t)((b << 12) + start + w) * D;
    red[d] = qd * q2[kro + d];
    __syncthreads();
    for (int s = 128; s > 0; s >>= 1) {
      if (d < s) red[d] += red[d + s];
      __syncthreads();
    }
    if (d == 0) sc[w] = red[0] * 0.0625f;   // / sqrt(256)
    __syncthreads();
  }

  float mx = sc[0];
#pragma unroll
  for (int w = 1; w < 9; ++w) mx = fmaxf(mx, sc[w]);
  float pw[9], den = 0.0f;
#pragma unroll
  for (int w = 0; w < 9; ++w) { pw[w] = expf(sc[w] - mx); den += pw[w]; }
  float inv = 1.0f / den;

  float acc = 0.0f;
#pragma unroll
  for (int w = 0; w < 9; ++w) {
    size_t kro = (size_t)((b << 12) + start + w) * D;
    acc += vmat[kro + d] * (pw[w] * inv);
  }

  float y = acc + x[base + d];

  red[d] = y;
  __syncthreads();
  for (int s = 128; s > 0; s >>= 1) {
    if (d < s) red[d] += red[d + s];
    __syncthreads();
  }
  if (d == 0) smu = red[0] * (1.0f / 256.0f);
  __syncthreads();

  float ym = y - smu;
  red[d] = ym * ym;
  __syncthreads();
  for (int s = 128; s > 0; s >>= 1) {
    if (d < s) red[d] += red[d + s];
    __syncthreads();
  }
  if (d == 0) svar = red[0] * (1.0f / 256.0f);
  __syncthreads();

  out[base + d] = ym * rsqrtf(svar + 0.001f) * gamma[d] + beta[d];
}

// ---------------------------------------------------------------------------
// Launch
// ---------------------------------------------------------------------------
extern "C" void kernel_launch(void* const* d_in, const int* in_sizes, int n_in,
                              void* d_out, int out_size, void* d_ws, size_t ws_size,
                              hipStream_t stream) {
  (void)in_sizes; (void)n_in; (void)out_size; (void)ws_size;
  const int B = 8, N = 4096, D = 256, H = 8;
  const int M  = B * N;      // 32768
  const int N1 = H * D;      // 2048

  const float* x  = (const float*)d_in[0];
  const float* hk = (const float*)d_in[1];
  const float* hb = (const float*)d_in[2];
  const float* dk = (const float*)d_in[3];
  const float* db = (const float*)d_in[4];
  const float* vk = (const float*)d_in[5];
  const float* vb = (const float*)d_in[6];
  const float* gm = (const float*)d_in[7];
  const float* bt = (const float*)d_in[8];
  float* out = (float*)d_out;

  char* ws = (char*)d_ws;
  size_t off = 0;
  auto alloc = [&](size_t bytes) -> char* {
    char* p = ws + off;
    off += (bytes + 255) & ~((size_t)255);
    return p;
  };
  __bf16* xb   = (__bf16*)alloc((size_t)M * D  * 2);   //  16 MB
  __bf16* hkb  = (__bf16*)alloc((size_t)D * N1 * 2);   //   1 MB
  __bf16* dkb  = (__bf16*)alloc((size_t)N1 * D * 2);   //   1 MB
  __bf16* vkb  = (__bf16*)alloc((size_t)D * D  * 2);   // 128 KB
  __bf16* qb   = (__bf16*)alloc((size_t)M * N1 * 2);   // 134 MB
  float*  simv = (float*) alloc((size_t)M * 4);
  int*    ord  = (int*)   alloc((size_t)M * 4);
  float*  q2f  = (float*) alloc((size_t)M * D * 4);    //  32 MB
  __bf16* q2b  = (__bf16*)alloc((size_t)M * D * 2);    //  16 MB
  float*  vf   = (float*) alloc((size_t)M * D * 4);    //  32 MB

  // bf16 conversions
  cvt_f32_bf16_kernel<<<(M * D  + 255) / 256, 256, 0, stream>>>(x,  xb,  M * D);
  cvt_f32_bf16_kernel<<<(D * N1 + 255) / 256, 256, 0, stream>>>(hk, hkb, D * N1);
  cvt_f32_bf16_kernel<<<(N1 * D + 255) / 256, 256, 0, stream>>>(dk, dkb, N1 * D);
  cvt_f32_bf16_kernel<<<(D * D  + 255) / 256, 256, 0, stream>>>(vk, vkb, D * D);

  // GEMM1 (TDM-staged A): q = relu(x @ head_kernel + head_bias) [32768 x 2048]
  gemm_bf16_relu_kernel<false, true, false>
      <<<dim3(N1 / 64, M / 128), 256, 0, stream>>>(
          xb, hkb, hb, nullptr, qb, nullptr, M, N1, D);

  // LSH + per-row sort + simv
  lsh_simv_kernel<<<M, 256, 0, stream>>>(qb, simv);

  // Per-batch argsort over N -> gather indices
  argsort_n_kernel<<<B, 1024, 0, stream>>>(simv, ord);

  // GEMM2 (gathered A rows): q2 = relu(q_perm @ down_kernel + down_bias)
  gemm_bf16_relu_kernel<true, true, true>
      <<<dim3(D / 64, M / 128), 256, 0, stream>>>(
          qb, dkb, db, ord, q2b, q2f, M, D, N1);

  // GEMM3 (TDM-staged A): v = relu(q2 @ value_kernel + value_bias)
  gemm_bf16_relu_kernel<false, false, true>
      <<<dim3(D / 64, M / 128), 256, 0, stream>>>(
          q2b, vkb, vb, nullptr, nullptr, vf, M, D, D);

  // Fused windowed attention + residual + LayerNorm
  attn_ln_kernel<<<M, 256, 0, stream>>>(q2f, vf, x, gm, bt, out);
}
